// MGDTModel_16535624089754
// MI455X (gfx1250) — matrix operand order repaired and checked by reading, not compiled, not run on hardware
//
#include <hip/hip_runtime.h>
#include <hip/hip_bf16.h>

// ---------------------------------------------------------------------------
// MGDT decision-transformer forward, CDNA5 (gfx1250, wave32, WMMA bf16).
//   B=8 T=32 S=39 L=1248 D=512 NH=8 dh=64 NL=6 DFF=2048
// GEMMs + attention matmuls: V_WMMA_F32_16X16X32_BF16.
// A-tiles staged into LDS by the Tensor Data Mover (tensor_load_to_lds +
// s_wait_tensorcnt), double-buffered; B = pre-transposed bf16 weights read
// with global_load_b128. Falls back to cooperative LDS staging if the TDM
// builtin is unavailable.
// ---------------------------------------------------------------------------

typedef unsigned short u16;
typedef unsigned int   u32;
typedef __attribute__((ext_vector_type(16))) __bf16 v16bf;
typedef __attribute__((ext_vector_type(8)))  float  v8f;
typedef __attribute__((ext_vector_type(4)))  u32    v4u;

#if defined(__AMDGCN__) && __has_builtin(__builtin_amdgcn_tensor_load_to_lds) && \
    __has_builtin(__builtin_amdgcn_s_wait_tensorcnt)
#define USE_TDM 1
#if __has_include(<hip/amd_detail/amd_gfx1250_TDM.h>)
#define TDM_SIX_ARGS 1
#endif
#endif

#ifdef USE_TDM
typedef __attribute__((ext_vector_type(4))) unsigned int tdm_v4u;
typedef __attribute__((ext_vector_type(8))) int          tdm_v8i;
typedef __attribute__((ext_vector_type(4))) int          tdm_v4i;
#endif

union FragBF { v16bf v; v4u q[2]; u32 u[8]; };

__device__ __forceinline__ u16 f2bf(float f) {
    u32 u = __float_as_uint(f);
    u32 r = u + 0x7FFFu + ((u >> 16) & 1u);   // round-to-nearest-even
    return (u16)(r >> 16);
}

__device__ __forceinline__ v8f wmma_bf16(const FragBF& a, const FragBF& b, v8f c) {
    return __builtin_amdgcn_wmma_f32_16x16x32_bf16(false, a.v, false, b.v,
                                                   (short)0, c, false, false);
}

// ---------------------------------------------------------------------------
// TDM: DMA a 64-row x 32-col bf16 tile (tile start gsrc, row stride K elems,
// tensor height M rows) into LDS at byte address ldsAddr. ISA 08 D# layout.
// ---------------------------------------------------------------------------
__device__ __forceinline__ void tdm_issue_tile(const u16* gsrc, u32 ldsAddr,
                                               int K, int M) {
#ifdef USE_TDM
    unsigned long long ga = (unsigned long long)(size_t)gsrc;
    tdm_v4u g0;
    g0[0] = 1u;                                          // count=1, load, user D#
    g0[1] = ldsAddr;                                     // lds_addr
    g0[2] = (u32)(ga & 0xFFFFFFFFu);                     // global_addr[31:0]
    g0[3] = (u32)((ga >> 32) & 0x01FFFFFFu) | (2u << 30);// global_addr[56:32], type=2
    u32 td0 = (u32)K, td1 = (u32)M;
    tdm_v8i g1;
    g1[0] = (int)(1u << 16);                             // mask=0, data_size=2B
    g1[1] = (int)((td0 & 0xFFFFu) << 16);                // tensor_dim0[15:0]
    g1[2] = (int)((td0 >> 16) | ((td1 & 0xFFFFu) << 16));// td0 hi, td1 lo
    g1[3] = (int)((td1 >> 16) | (32u << 16));            // td1 hi, tile_dim0=32
    g1[4] = (int)64u;                                    // tile_dim1=64, tile_dim2=0
    g1[5] = (int)td0;                                    // tensor_dim0_stride[31:0]
    g1[6] = 0;                                           // stride hi, dim1_stride lo
    g1[7] = 0;
    tdm_v4i gz = {0, 0, 0, 0};
#ifdef TDM_SIX_ARGS
    tdm_v8i z8 = {0, 0, 0, 0, 0, 0, 0, 0};
    __builtin_amdgcn_tensor_load_to_lds(g0, g1, gz, gz, z8, 0);
#else
    __builtin_amdgcn_tensor_load_to_lds(g0, g1, gz, gz, 0);
#endif
#else
    (void)gsrc; (void)ldsAddr; (void)K; (void)M;
#endif
}

__device__ __forceinline__ void tdm_wait() {
#ifdef USE_TDM
    __builtin_amdgcn_s_wait_tensorcnt(0);
#endif
}

// ---------------------------------------------------------------------------
// Weight prep: f32 (K x N) -> bf16 transposed (N x Kdst), zero-padded K tail.
// ---------------------------------------------------------------------------
__global__ void cvt_transpose(const float* __restrict__ src, u16* __restrict__ dst,
                              int Ksrc, int Kdst, int N) {
    size_t idx = (size_t)blockIdx.x * blockDim.x + threadIdx.x;
    if (idx >= (size_t)N * Kdst) return;
    int k = (int)(idx % Kdst);
    int n = (int)(idx / Kdst);
    dst[idx] = (k < Ksrc) ? f2bf(src[(size_t)k * N + n]) : (u16)0;
}

// ---------------------------------------------------------------------------
// Patch gather: frames (B,T,C,84,84) -> X (9216 x 800) bf16, cols 784..799 = 0
// ---------------------------------------------------------------------------
__global__ void gather_patches(const float* __restrict__ frames, u16* __restrict__ X) {
    size_t idx = (size_t)blockIdx.x * blockDim.x + threadIdx.x;
    if (idx >= (size_t)9216 * 800) return;
    int col  = (int)(idx % 800);
    int rowi = (int)(idx / 800);
    if (col >= 784) { X[idx] = 0; return; }
    int gx = rowi % 6;
    int gy = (rowi / 6) % 6;
    int bt = rowi / 36;                    // b*T + t
    int px = col % 14;
    int py = (col / 14) % 14;
    int c  = col / 196;
    size_t fidx = (((size_t)bt * 4 + c) * 84 + (gy * 14 + py)) * 84 + (gx * 14 + px);
    X[idx] = f2bf(frames[fidx]);
}

// ---------------------------------------------------------------------------
// WMMA bf16 GEMM: C[M,N] = A[M,K] * Bt[N,K]^T (+bias)(+GELU)(+residual)
// Block = 256 threads = 8 waves; block tile 64x128; wave tile 32x32 (2x2 WMMA).
// A block-tile (64x32) staged into double-buffered LDS by the TDM; all waves
// read A-fragments with ds_load_b128. B-fragments: two global_load_b128 each.
// M%64==0, N%128==0, K%32==0 (compile-time). lda = ldb = K.
// ---------------------------------------------------------------------------
template<int K, bool GELU_, bool RES_, bool OUTF_, bool OUTB_>
__global__ __launch_bounds__(256) void gemm_wmma(
    const u16* __restrict__ A, const u16* __restrict__ Bt,
    const float* __restrict__ bias, const float* __restrict__ residual,
    float* __restrict__ outF, u16* __restrict__ outB, int N, int M)
{
    __shared__ u16 tileA[2][64 * 32];                  // 2 x 4KB double buffer

    const int lane = threadIdx.x & 31;
    const int wave = threadIdx.x >> 5;
    const int ln   = lane & 15;
    const int hi   = lane >> 4;
    const int wm   = wave >> 2;                        // 0..1 (M)
    const int wn   = wave & 3;                         // 0..3 (N)
    const int m0   = blockIdx.x * 64  + wm * 32;
    const int n0   = blockIdx.y * 128 + wn * 32;

    const u16* aBlk = A + (size_t)blockIdx.x * 64 * K; // block's A rows
    const u16* b0 = Bt + (size_t)(n0 + ln) * K;        // col for nt=0
    const u16* b1 = b0 + (size_t)16 * K;               // col for nt=1
    const int koffA = hi * 16;                         // A-frag lane-half offset
    const int koffB = hi * 16;                         // B-frag lane-half offset
    const bool issuer = (wave == 0);

    __builtin_prefetch(b0, 0, 3);

#ifdef USE_TDM
    if (issuer) tdm_issue_tile(aBlk, (u32)(size_t)&tileA[0][0], K, M);
#endif

    v8f z = {};
    v8f acc00 = z, acc01 = z, acc10 = z, acc11 = z;

    int buf = 0;
    for (int kb = 0; kb < K; kb += 32) {
#ifdef USE_TDM
        if (issuer) tdm_wait();                        // current buffer ready
        __syncthreads();
        if (issuer && (kb + 32 < K))                   // overlap next DMA
            tdm_issue_tile(aBlk + kb + 32, (u32)(size_t)&tileA[buf ^ 1][0], K, M);
#else
        {   // cooperative fallback staging: 256 threads x 16B
            int t = threadIdx.x;
            int row = t >> 2, cc = (t & 3) * 8;
            *(v4u*)&tileA[buf][row * 32 + cc] =
                *(const v4u*)(aBlk + (size_t)row * K + kb + cc);
            __syncthreads();
        }
#endif
        FragBF fa0, fa1, fb0, fb1;
        const u16* ar0 = &tileA[buf][(wm * 32 + ln) * 32];
        const u16* ar1 = ar0 + 16 * 32;
        fa0.q[0] = *(const v4u*)(ar0 + koffA);
        fa0.q[1] = *(const v4u*)(ar0 + 8 + koffA);
        fa1.q[0] = *(const v4u*)(ar1 + koffA);
        fa1.q[1] = *(const v4u*)(ar1 + 8 + koffA);
        fb0.q[0] = *(const v4u*)(b0 + kb + koffB);
        fb0.q[1] = *(const v4u*)(b0 + kb + koffB + 8);
        fb1.q[0] = *(const v4u*)(b1 + kb + koffB);
        fb1.q[1] = *(const v4u*)(b1 + kb + koffB + 8);
        acc00 = wmma_bf16(fa0, fb0, acc00);
        acc01 = wmma_bf16(fa0, fb1, acc01);
        acc10 = wmma_bf16(fa1, fb0, acc10);
        acc11 = wmma_bf16(fa1, fb1, acc11);
        __syncthreads();                               // done reading this buffer
        buf ^= 1;
    }

    v8f* accs[4] = { &acc00, &acc01, &acc10, &acc11 };
#pragma unroll
    for (int mt = 0; mt < 2; ++mt)
#pragma unroll
        for (int nt = 0; nt < 2; ++nt) {
            v8f& acc = *accs[mt * 2 + nt];
#pragma unroll
            for (int r = 0; r < 8; ++r) {
                int row = m0 + mt * 16 + r + 8 * hi;
                int col = n0 + nt * 16 + ln;
                float v = acc[r] + bias[col];
                if constexpr (GELU_) v = 0.5f * v * (1.0f + erff(v * 0.70710678118654752f));
                size_t o = (size_t)row * N + col;
                if constexpr (RES_)  v += residual[o];
                if constexpr (OUTF_) outF[o] = v;
                if constexpr (OUTB_) outB[o] = f2bf(v);
            }
        }
}

// ---------------------------------------------------------------------------
// Sequence assembly: h[b,l,:] = token + pos_E[l,:] + type_E[type,:]
// ---------------------------------------------------------------------------
__global__ __launch_bounds__(256) void assemble_seq(
    const float* __restrict__ obs_tok,  // (B*T*36, 512) f32
    const float* __restrict__ retE, const float* __restrict__ actE,
    const float* __restrict__ rewE, const float* __restrict__ posE,
    const float* __restrict__ typeE,
    const int* __restrict__ rtg, const int* __restrict__ act,
    const int* __restrict__ rew,
    float* __restrict__ h)
{
    int l = blockIdx.x % 1248;
    int b = blockIdx.x / 1248;
    int t = l / 39, off = l % 39;
    int bt = b * 32 + t;
#pragma unroll
    for (int rep = 0; rep < 2; ++rep) {
        int d = threadIdx.x + rep * 256;
        float tok; int type;
        if (off < 36)       { tok = obs_tok[((size_t)bt * 36 + off) * 512 + d]; type = 0; }
        else if (off == 36) { tok = retE[(size_t)rtg[bt] * 512 + d];            type = 1; }
        else if (off == 37) { tok = actE[(size_t)act[bt] * 512 + d];            type = 2; }
        else                { tok = rewE[(size_t)rew[bt] * 512 + d];            type = 3; }
        h[((size_t)b * 1248 + l) * 512 + d] = tok + posE[(size_t)l * 512 + d]
                                                  + typeE[(size_t)type * 512 + d];
    }
}

// ---------------------------------------------------------------------------
// LayerNorm (D=512) f32 in -> bf16 out. 128 threads per row.
// ---------------------------------------------------------------------------
__global__ __launch_bounds__(128) void ln_bf16(
    const float* __restrict__ x, const float* __restrict__ g,
    const float* __restrict__ bb, u16* __restrict__ out)
{
    int row = blockIdx.x;
    const float* xr = x + (size_t)row * 512;
    float v[4]; float s = 0.f, s2 = 0.f;
#pragma unroll
    for (int i = 0; i < 4; ++i) {
        v[i] = xr[threadIdx.x * 4 + i];
        s += v[i]; s2 += v[i] * v[i];
    }
    for (int d = 16; d > 0; d >>= 1) { s += __shfl_xor(s, d); s2 += __shfl_xor(s2, d); }
    __shared__ float red[8];
    int w = threadIdx.x >> 5, lane = threadIdx.x & 31;
    if (lane == 0) { red[w] = s; red[4 + w] = s2; }
    __syncthreads();
    s  = red[0] + red[1] + red[2] + red[3];
    s2 = red[4] + red[5] + red[6] + red[7];
    float mean = s * (1.0f / 512.0f);
    float var  = s2 * (1.0f / 512.0f) - mean * mean;
    float inv  = rsqrtf(var + 1e-5f);
#pragma unroll
    for (int i = 0; i < 4; ++i) {
        int c = threadIdx.x * 4 + i;
        out[(size_t)row * 512 + c] = f2bf((v[i] - mean) * inv * g[c] + bb[c]);
    }
}

// ---------------------------------------------------------------------------
// Flash attention, 1 wave per (b, head, 16-query tile).
// qkv bf16: (B, L, 1536) with [q|k|v] each 512 wide; mask = causal OR
// same-step obs<->obs. Scores stay in registers; P restaged via 1KB LDS.
// ---------------------------------------------------------------------------
__global__ __launch_bounds__(32) void attn_flash(
    const u16* __restrict__ qkv, u16* __restrict__ attn_out)
{
    __shared__ u16 ldsP[16][32];
    const int lane = threadIdx.x;
    const int ln = lane & 15, hi = lane >> 4;
    int bid = blockIdx.x;
    int qt = bid % 78;
    int hh = (bid / 78) & 7;
    int b  = bid / (78 * 8);
    int q0 = qt * 16;
    const size_t qkvB = (size_t)b * 1248 * 1536;
    const int koffA = hi * 16;
    const int koffB = hi * 16;

    FragBF qf[2];
    {
        const u16* qp = qkv + qkvB + (size_t)(q0 + ln) * 1536 + hh * 64;
#pragma unroll
        for (int c = 0; c < 2; ++c) {
            qf[c].q[0] = *(const v4u*)(qp + c * 32 + koffA);
            qf[c].q[1] = *(const v4u*)(qp + c * 32 + 8 + koffA);
        }
    }

    v8f z = {};
    v8f o[4]; for (int i = 0; i < 4; ++i) o[i] = z;
    float mrow[8], lrow[8];
#pragma unroll
    for (int r = 0; r < 8; ++r) { mrow[r] = -1e30f; lrow[r] = 0.f; }

    int q_hi = q0 + 15;
    int st = q_hi / 39;
    int lim = st * 39 + 35; if (q_hi > lim) lim = q_hi;   // causal + same-step obs window
    int niter = (lim + 32) / 32;

    for (int it = 0; it < niter; ++it) {
        int k0 = it * 32;
        FragBF kf[2][2];
#pragma unroll
        for (int nt = 0; nt < 2; ++nt) {
            const u16* kp = qkv + qkvB + (size_t)(k0 + nt * 16 + ln) * 1536 + 512 + hh * 64;
#pragma unroll
            for (int c = 0; c < 2; ++c) {
                kf[nt][c].q[0] = *(const v4u*)(kp + c * 32 + koffB);
                kf[nt][c].q[1] = *(const v4u*)(kp + c * 32 + koffB + 8);
            }
        }
        v8f s0 = z, s1 = z;
        s0 = wmma_bf16(qf[0], kf[0][0], s0);
        s0 = wmma_bf16(qf[1], kf[0][1], s0);
        s1 = wmma_bf16(qf[0], kf[1][0], s1);
        s1 = wmma_bf16(qf[1], kf[1][1], s1);

        float p0[8], p1[8], alpha[8];
#pragma unroll
        for (int r = 0; r < 8; ++r) {
            int qrow = q0 + r + 8 * hi;
            int kc0 = k0 + ln, kc1 = k0 + 16 + ln;
            int  qstep = qrow / 39;
            bool obsq  = (qrow % 39) < 36;
            bool a0 = (kc0 <= qrow) || ((kc0 / 39) == qstep && obsq && (kc0 % 39) < 36);
            bool a1 = (kc1 <= qrow) || ((kc1 / 39) == qstep && obsq && (kc1 % 39) < 36);
            float v0 = s0[r] * 0.125f + (a0 ? 0.f : -1e9f);
            float v1 = s1[r] * 0.125f + (a1 ? 0.f : -1e9f);
            float mx = fmaxf(v0, v1);
            mx = fmaxf(mx, __shfl_xor(mx, 1));
            mx = fmaxf(mx, __shfl_xor(mx, 2));
            mx = fmaxf(mx, __shfl_xor(mx, 4));
            mx = fmaxf(mx, __shfl_xor(mx, 8));
            float mnew = fmaxf(mrow[r], mx);
            float al = __expf(mrow[r] - mnew);
            float e0 = __expf(v0 - mnew), e1 = __expf(v1 - mnew);
            float sum = e0 + e1;
            sum += __shfl_xor(sum, 1);
            sum += __shfl_xor(sum, 2);
            sum += __shfl_xor(sum, 4);
            sum += __shfl_xor(sum, 8);
            lrow[r] = lrow[r] * al + sum;
            mrow[r] = mnew;
            alpha[r] = al;
            p0[r] = e0; p1[r] = e1;
        }
#pragma unroll
        for (int nt2 = 0; nt2 < 4; ++nt2)
#pragma unroll
            for (int r = 0; r < 8; ++r) o[nt2][r] *= alpha[r];

#pragma unroll
        for (int r = 0; r < 8; ++r) {
            int m = r + 8 * hi;
            ldsP[m][ln]      = f2bf(p0[r]);
            ldsP[m][16 + ln] = f2bf(p1[r]);
        }
        __builtin_amdgcn_wave_barrier();
        FragBF pf;
        pf.q[0] = *(const v4u*)&ldsP[ln][koffA];
        pf.q[1] = *(const v4u*)&ldsP[ln][8 + koffA];
        __builtin_amdgcn_wave_barrier();

#pragma unroll
        for (int nt2 = 0; nt2 < 4; ++nt2) {
            FragBF vf;
            int dn = 1024 + hh * 64 + nt2 * 16 + ln;
#pragma unroll
            for (int j = 0; j < 8; ++j) {
                int key0 = k0 + koffB + 2 * j;
                u32 lo  = qkv[qkvB + (size_t)key0       * 1536 + dn];
                u32 h16 = qkv[qkvB + (size_t)(key0 + 1) * 1536 + dn];
                vf.u[j] = lo | (h16 << 16);
            }
            o[nt2] = wmma_bf16(pf, vf, o[nt2]);
        }
    }

#pragma unroll
    for (int nt2 = 0; nt2 < 4; ++nt2)
#pragma unroll
        for (int r = 0; r < 8; ++r) {
            int row = q0 + r + 8 * hi;
            float val = o[nt2][r] / lrow[r];
            attn_out[((size_t)b * 1248 + row) * 512 + hh * 64 + nt2 * 16 + ln] = f2bf(val);
        }
}

// ---------------------------------------------------------------------------
// Output heads (tiny GEMMs, f32 VALU): 256 rows each x {120,18,3}
// ---------------------------------------------------------------------------
__global__ __launch_bounds__(256) void heads_kernel(
    const float* __restrict__ h,
    const float* __restrict__ Wr, const float* __restrict__ br,
    const float* __restrict__ Wa, const float* __restrict__ ba,
    const float* __restrict__ Ww, const float* __restrict__ bw,
    float* __restrict__ out)
{
    int idx = blockIdx.x * blockDim.x + threadIdx.x;
    if (idx >= 36096) return;
    int N, loff, obase, r, n; const float *W, *bias;
    if (idx < 30720)      { N = 120; loff = 35; obase = 0;     W = Wr; bias = br; r = idx / 120;           n = idx % 120; }
    else if (idx < 35328) { N = 18;  loff = 36; obase = 30720; W = Wa; bias = ba; r = (idx - 30720) / 18;  n = (idx - 30720) % 18; }
    else                  { N = 3;   loff = 37; obase = 35328; W = Ww; bias = bw; r = (idx - 35328) / 3;   n = (idx - 35328) % 3; }
    int b = r / 32, t = r % 32;
    int l = t * 39 + loff;
    const float* hr = h + ((size_t)b * 1248 + l) * 512;
    float acc = bias[n];
    for (int d = 0; d < 512; ++d) acc += hr[d] * W[(size_t)d * N + n];
    out[obase + r * N + n] = acc;
}

// ---------------------------------------------------------------------------
// Host-side orchestration
// ---------------------------------------------------------------------------
static inline size_t al256(size_t x) { return (x + 255) & ~(size_t)255; }

extern "C" void kernel_launch(void* const* d_in, const int* in_sizes, int n_in,
                              void* d_out, int out_size, void* d_ws, size_t ws_size,
                              hipStream_t stream) {
    (void)in_sizes; (void)n_in; (void)out_size; (void)ws_size;
    const float* frames   = (const float*)d_in[0];
    const int*   rtg      = (const int*)  d_in[1];
    const int*   actions  = (const int*)  d_in[2];
    const int*   rewards  = (const int*)  d_in[3];
    const float* patch_W  = (const float*)d_in[4];
    const float* patch_b  = (const float*)d_in[5];
    const float* return_E = (const float*)d_in[6];
    const float* action_E = (const float*)d_in[7];
    const float* reward_E = (const float*)d_in[8];
    const float* pos_E    = (const float*)d_in[9];
    const float* type_E   = (const float*)d_in[10];
    const float* Wqkv     = (const float*)d_in[11];
    const float* bqkv     = (const float*)d_in[12];
    const float* Wo       = (const float*)d_in[13];
    const float* bo       = (const float*)d_in[14];
    const float* ln1_g    = (const float*)d_in[15];
    const float* ln1_b    = (const float*)d_in[16];
    const float* ln2_g    = (const float*)d_in[17];
    const float* ln2_b    = (const float*)d_in[18];
    const float* W1       = (const float*)d_in[19];
    const float* b1       = (const float*)d_in[20];
    const float* W2       = (const float*)d_in[21];
    const float* b2       = (const float*)d_in[22];
    const float* hW_ret   = (const float*)d_in[23];
    const float* hb_ret   = (const float*)d_in[24];
    const float* hW_act   = (const float*)d_in[25];
    const float* hb_act   = (const float*)d_in[26];
    const float* hW_rew   = (const float*)d_in[27];
    const float* hb_rew   = (const float*)d_in[28];
    float* out = (float*)d_out;

    const int BB = 8, L = 1248, D = 512, ML = BB * L;   // 9984 rows
    char* p = (char*)d_ws; size_t off = 0;
    auto carve = [&](size_t bytes) { char* q = p + off; off = al256(off + bytes); return q; };
    u16*   wPW   = (u16*)  carve((size_t)512 * 800 * 2);         // patch_W^T, padded
    u16*   wQKV  = (u16*)  carve(6ull * 1536 * 512 * 2);         // Wqkv^T
    u16*   wWO   = (u16*)  carve(6ull * 512 * 512 * 2);          // Wo^T
    u16*   wW1   = (u16*)  carve(6ull * 2048 * 512 * 2);         // W1^T
    u16*   wW2   = (u16*)  carve(6ull * 512 * 2048 * 2);         // W2^T
    u16*   Xp    = (u16*)  carve((size_t)9216 * 800 * 2);        // patch matrix, padded
    float* obsT  = (float*)carve((size_t)9216 * 512 * 4);
    float* h     = (float*)carve((size_t)ML * D * 4);
    u16*   x_ln  = (u16*)  carve((size_t)ML * D * 2);
    u16*   qkvB  = (u16*)  carve((size_t)ML * 1536 * 2);
    u16*   attnO = (u16*)  carve((size_t)ML * D * 2);
    u16*   hidB  = (u16*)  carve((size_t)ML * 2048 * 2);

    auto cvtT = [&](const float* src, u16* dst, int Ksrc, int Kdst, int N) {
        size_t n = (size_t)N * Kdst;
        cvt_transpose<<<(unsigned)((n + 255) / 256), 256, 0, stream>>>(src, dst, Ksrc, Kdst, N);
    };
    cvtT(patch_W, wPW, 784, 800, 512);
    for (int i = 0; i < 6; ++i) {
        cvtT(Wqkv + (size_t)i * 512 * 1536, wQKV + (size_t)i * 1536 * 512, 512, 512, 1536);
        cvtT(Wo   + (size_t)i * 512 * 512,  wWO  + (size_t)i * 512 * 512,  512, 512, 512);
        cvtT(W1   + (size_t)i * 512 * 2048, wW1  + (size_t)i * 2048 * 512, 512, 512, 2048);
        cvtT(W2   + (size_t)i * 2048 * 512, wW2  + (size_t)i * 512 * 2048, 2048, 2048, 512);
    }

    gather_patches<<<(unsigned)(((size_t)9216 * 800 + 255) / 256), 256, 0, stream>>>(frames, Xp);

    // Patch embedding: (9216 x 800) x (800 x 512) + patch_b -> obs tokens f32
    gemm_wmma<800, false, false, true, false>
        <<<dim3(9216 / 64, 512 / 128), 256, 0, stream>>>(Xp, wPW, patch_b, nullptr,
                                                         obsT, nullptr, 512, 9216);

    // Token assembly + pos/type embeddings -> residual stream h
    assemble_seq<<<BB * L, 256, 0, stream>>>(obsT, return_E, action_E, reward_E,
                                             pos_E, type_E, rtg, actions, rewards, h);

    for (int i = 0; i < 6; ++i) {
        // ---- attention block ----
        ln_bf16<<<ML, 128, 0, stream>>>(h, ln1_g + i * 512, ln1_b + i * 512, x_ln);
        gemm_wmma<512, false, false, false, true>
            <<<dim3(ML / 64, 1536 / 128), 256, 0, stream>>>(
                x_ln, wQKV + (size_t)i * 1536 * 512, bqkv + (size_t)i * 1536,
                nullptr, nullptr, qkvB, 1536, ML);
        attn_flash<<<BB * 8 * 78, 32, 0, stream>>>(qkvB, attnO);
        gemm_wmma<512, false, true, true, false>
            <<<dim3(ML / 64, 512 / 128), 256, 0, stream>>>(
                attnO, wWO + (size_t)i * 512 * 512, bo + (size_t)i * 512,
                h, h, nullptr, 512, ML);
        // ---- MLP block ----
        ln_bf16<<<ML, 128, 0, stream>>>(h, ln2_g + i * 512, ln2_b + i * 512, x_ln);
        gemm_wmma<512, true, false, false, true>
            <<<dim3(ML / 64, 2048 / 128), 256, 0, stream>>>(
                x_ln, wW1 + (size_t)i * 2048 * 512, b1 + (size_t)i * 2048,
                nullptr, nullptr, hidB, 2048, ML);
        gemm_wmma<2048, false, true, true, false>
            <<<dim3(ML / 64, 512 / 128), 256, 0, stream>>>(
                hidB, wW2 + (size_t)i * 512 * 2048, b2 + (size_t)i * 512,
                h, h, nullptr, 512, ML);
    }

    heads_kernel<<<(36096 + 255) / 256, 256, 0, stream>>>(
        h, hW_ret, hb_ret, hW_act, hb_act, hW_rew, hb_rew, out);
}
